// IOCClassifier_18030272708865
// MI455X (gfx1250) — compile-verified, matrix operand order checked by reading, not run on hardware
//
#include <hip/hip_runtime.h>
#include <hip/hip_bf16.h>

typedef __bf16 v16bf __attribute__((ext_vector_type(16)));
typedef __bf16 v8bf  __attribute__((ext_vector_type(8)));
typedef __bf16 v2bf  __attribute__((ext_vector_type(2)));
typedef float  v8f   __attribute__((ext_vector_type(8)));
typedef int    v4i   __attribute__((ext_vector_type(4)));

#define HID  256
#define FIN  128
#define RREL 6
#define NBAS 8
#define KCAT 1792   /* RREL*HID + HID */
#define LN_EPS 1e-5f

#if __has_builtin(__builtin_amdgcn_global_load_async_to_lds_b128)
#define ASYNC_OK 1
#else
#define ASYNC_OK 0
#endif

union Frag16 { v16bf v; v8bf h[2]; unsigned short s[16]; unsigned int u[8]; };

// native f32 -> bf16 (RTNE) via hardware convert
static __device__ __forceinline__ unsigned short f2bf(float f) {
  union { __bf16 b; unsigned short s; } u;
  u.b = (__bf16)f;
  return u.s;
}
static __device__ __forceinline__ unsigned int f2bf2(float a, float b) {
  union { v2bf v; unsigned int u; } t;
  t.v = (v2bf){(__bf16)a, (__bf16)b};
  return t.u;
}

static __device__ __forceinline__ void wait_async0() {
#if ASYNC_OK
#if __has_builtin(__builtin_amdgcn_s_wait_asynccnt)
  __builtin_amdgcn_s_wait_asynccnt(0);
#else
  asm volatile("s_wait_asynccnt 0" ::: "memory");
#endif
#endif
}

// B fragment from pre-swizzled weights: layout [ktile][ntile][lane][16 elems]
static __device__ __forceinline__ v16bf load_bfrag(const unsigned short* __restrict__ w,
                                                   int ktile, int ntile, int lane, int ntiles) {
  const unsigned short* p = w + ((((size_t)ktile * ntiles + ntile) * 32 + lane) * 16);
  Frag16 u;
  u.h[0] = *(const v8bf*)(p);
  u.h[1] = *(const v8bf*)(p + 8);
  return u.v;
}

// ---------------- weight preparation kernels ----------------

__global__ void make_wcat_kernel(const float* __restrict__ comp,
                                 const float* __restrict__ basis,
                                 const float* __restrict__ root,
                                 unsigned short* __restrict__ wsw) {
  int g = blockIdx.x * 256 + threadIdx.x;           // KCAT blocks -> KCAT*HID elems
  int j    = g & 15;
  int lane = (g >> 4) & 31;
  int nt   = (g >> 9) & 15;
  int kt   = g >> 13;                                // 0..55
  int K    = kt * 32 + ((lane >> 4) << 3) + (j < 8 ? j : j + 8);
  int col  = nt * 16 + (lane & 15);
  float val;
  if (K < RREL * HID) {
    int r = K >> 8, i = K & 255;
    float s = 0.f;
#pragma unroll
    for (int b = 0; b < NBAS; ++b)
      s += comp[r * NBAS + b] * basis[((size_t)b * HID + i) * HID + col];
    val = s;
  } else {
    val = root[(size_t)(K - RREL * HID) * HID + col];
  }
  wsw[g] = f2bf(val);
}

__global__ void make_projw_kernel(const float* __restrict__ projW,
                                  unsigned short* __restrict__ pw) {
  int g = blockIdx.x * 256 + threadIdx.x;           // 512 blocks
  int j    = g & 15;
  int lane = (g >> 4) & 31;
  int nt   = (g >> 9) & 15;
  int kt   = (g >> 13) & 3;
  int t    = g >> 15;
  int K    = kt * 32 + ((lane >> 4) << 3) + (j < 8 ? j : j + 8);
  int col  = nt * 16 + (lane & 15);
  pw[g] = f2bf(projW[((size_t)t * FIN + K) * HID + col]);
}

__global__ void relw_kernel(const float* __restrict__ ete,
                            const float* __restrict__ emlpW,
                            float* __restrict__ relw) {
  int g = blockIdx.x * 256 + threadIdx.x;           // RREL blocks
  int r = g >> 8, c = g & 255;
  float s = 0.f;
#pragma unroll
  for (int k = 0; k < 16; ++k) s += ete[r * 16 + k] * emlpW[k * HID + c];
  relw[g] = s;
}

// ---------------- edge kernels ----------------

__global__ void edge_bias_kernel(const int* __restrict__ edst, const int* __restrict__ etyp,
                                 const float* __restrict__ ew, const float* __restrict__ relw,
                                 const float* __restrict__ emlpW, const float* __restrict__ emlpb,
                                 float* __restrict__ agg, float* __restrict__ deg,
                                 float* __restrict__ cnt) {
  int e = blockIdx.x, c = threadIdx.x;
  int d = edst[e], r = etyp[e];
  float w = ew[e];
  float v = relw[r * HID + c] + w * emlpW[16 * HID + c] + emlpb[c];
  v = fmaxf(v, 0.f);
  atomicAdd(&agg[(size_t)d * HID + c], v);
  if (c == 0) {
    atomicAdd(&deg[d], 1.f);
    atomicAdd(&cnt[d * RREL + r], 1.f);
  }
}

__global__ void finalize_bias_kernel(float* __restrict__ ebias,
                                     const float* __restrict__ deg, int n) {
  int i = blockIdx.x * 256 + threadIdx.x;
  if (i < n) ebias[i] = 0.1f * ebias[i] / fmaxf(deg[i >> 8], 1.f);
}

__global__ void inv_cnt_kernel(float* __restrict__ cnt, int n) {
  int i = blockIdx.x * 256 + threadIdx.x;
  if (i < n) cnt[i] = 1.f / fmaxf(cnt[i], 1.f);
}

__global__ void scatter_kernel(const int* __restrict__ esrc, const int* __restrict__ edst,
                               const int* __restrict__ etyp, const float* __restrict__ h,
                               float* __restrict__ sums) {
  int e = blockIdx.x, c = threadIdx.x;
  size_t key = (size_t)edst[e] * RREL + etyp[e];
  atomicAdd(&sums[key * HID + c], h[(size_t)esrc[e] * HID + c]);
}

// 2 elements per thread -> packed v_cvt_pk_bf16_f32
__global__ void scale_mean_kernel(const float* __restrict__ sums,
                                  const float* __restrict__ invc,
                                  unsigned int* __restrict__ mean2, size_t n2) {
  size_t i = (size_t)blockIdx.x * 256 + threadIdx.x;
  if (i >= n2) return;
  size_t e = i * 2;
  float inv = invc[e >> 8];           // e even => e and e+1 share a segment
  float2 s = *(const float2*)(sums + e);
  mean2[i] = f2bf2(s.x * inv, s.y * inv);
}

// ---------------- fused projection: x@W + b -> LN -> relu ----------------

__global__ __launch_bounds__(256) void proj_kernel(
    const float* __restrict__ x0, const float* __restrict__ x1,
    const float* __restrict__ x2, const float* __restrict__ x3,
    int n0, int n1, int n2,
    const unsigned short* __restrict__ pwsw,
    const float* __restrict__ pbias, const float* __restrict__ pg,
    const float* __restrict__ pb, float* __restrict__ hout) {
  __shared__ float tile[16 * HID];
  __shared__ float s_mu[16], s_rs[16];
  int tid = threadIdx.x, wave = tid >> 5, lane = tid & 31;
  int rows0 = blockIdx.x * 16;
  int t, start;
  if (rows0 < n0)                { t = 0; start = 0; }
  else if (rows0 < n0 + n1)      { t = 1; start = n0; }
  else if (rows0 < n0 + n1 + n2) { t = 2; start = n0 + n1; }
  else                           { t = 3; start = n0 + n1 + n2; }
  const float* xs = (t == 0 ? x0 : t == 1 ? x1 : t == 2 ? x2 : x3);
  const float* xrow = xs + (size_t)(rows0 - start + (lane & 15)) * FIN;
  const unsigned short* pw = pwsw + (size_t)t * (4 * 16 * 32 * 16);

  v8f acc0 = {}, acc1 = {};
  int nt0 = wave * 2, nt1 = wave * 2 + 1;
#pragma unroll
  for (int step = 0; step < 4; ++step) {
    Frag16 a;
    int kb = step * 32 + ((lane >> 4) << 3);
    float4 q0 = *(const float4*)(xrow + kb);
    float4 q1 = *(const float4*)(xrow + kb + 4);
    float4 q2 = *(const float4*)(xrow + kb + 16);
    float4 q3 = *(const float4*)(xrow + kb + 20);
    a.u[0] = f2bf2(q0.x, q0.y); a.u[1] = f2bf2(q0.z, q0.w);
    a.u[2] = f2bf2(q1.x, q1.y); a.u[3] = f2bf2(q1.z, q1.w);
    a.u[4] = f2bf2(q2.x, q2.y); a.u[5] = f2bf2(q2.z, q2.w);
    a.u[6] = f2bf2(q3.x, q3.y); a.u[7] = f2bf2(q3.z, q3.w);
    v16bf b0 = load_bfrag(pw, step, nt0, lane, 16);
    v16bf b1 = load_bfrag(pw, step, nt1, lane, 16);
    acc0 = __builtin_amdgcn_wmma_f32_16x16x32_bf16(false, a.v, false, b0, (short)0, acc0, false, false);
    acc1 = __builtin_amdgcn_wmma_f32_16x16x32_bf16(false, a.v, false, b1, (short)0, acc1, false, false);
  }

  const float* bias = pbias + t * HID;
  const float* gg = pg + t * HID;
  const float* bb = pb + t * HID;
  int hi = lane >> 4, cl = lane & 15;
#pragma unroll
  for (int v = 0; v < 8; ++v) {
    int r = v + 8 * hi;
    int c0 = wave * 32 + cl, c1 = c0 + 16;
    tile[r * HID + c0] = acc0[v] + bias[c0];
    tile[r * HID + c1] = acc1[v] + bias[c1];
  }
  __syncthreads();
#pragma unroll
  for (int rr = 0; rr < 2; ++rr) {
    int r = wave * 2 + rr;
    float s = 0.f, sq = 0.f;
    for (int c = lane; c < HID; c += 32) { float xv = tile[r * HID + c]; s += xv; sq += xv * xv; }
#pragma unroll
    for (int o = 16; o > 0; o >>= 1) { s += __shfl_xor(s, o, 32); sq += __shfl_xor(sq, o, 32); }
    if (lane == 0) {
      float m = s / 256.f;
      s_mu[r] = m;
      s_rs[r] = rsqrtf(fmaxf(sq / 256.f - m * m, 0.f) + LN_EPS);
    }
  }
  __syncthreads();
  for (int r = 0; r < 16; ++r) {
    float xv = tile[r * HID + tid];
    float y = (xv - s_mu[r]) * s_rs[r] * gg[tid] + bb[tid];
    hout[(size_t)(rows0 + r) * HID + tid] = fmaxf(y, 0.f);
  }
}

// ---------------- fused RGCN layer: [mean|h]@Wcat + bias + ebias -> LN -> relu -> +h ----------------

__global__ __launch_bounds__(256) void rgcn_layer_kernel(
    const unsigned short* __restrict__ mean,   // [N,1536] bf16
    const float* __restrict__ hin,             // [N,256]
    const unsigned short* __restrict__ wsw,    // swizzled [56][16][32][16] bf16
    const float* __restrict__ cbias, const float* __restrict__ g,
    const float* __restrict__ b, const float* __restrict__ ebias,
    float* __restrict__ hout) {
  __shared__ unsigned short abuf[2][16 * 32];
  __shared__ float tile[16 * HID];
  __shared__ float s_mu[16], s_rs[16];
  int tid = threadIdx.x, wave = tid >> 5, lane = tid & 31;
  int rows0 = blockIdx.x * 16;

  // manual staging (f32 h tail): thread loads one u32 (2 bf16) per step
  int arow = tid >> 4;
  int ak = (tid & 15) * 2;
  size_t grow = (size_t)rows0 + arow;

  auto stage_h = [&](int step) -> unsigned int {    // steps 48..55
    int k2 = step * 32 + ak - RREL * HID;
    float2 f = *(const float2*)(hin + grow * HID + k2);
    return f2bf2(f.x, f.y);
  };
  auto stage_mean = [&](int step) -> unsigned int { // steps 0..47 (fallback)
    return *(const unsigned int*)(mean + grow * (RREL * HID) + step * 32 + ak);
  };

#if ASYNC_OK
  // DMA staging: 64 threads issue one B128 (16B) each -> full 16x32 bf16 tile
  auto async_issue = [&](int step, unsigned short* dst) {
    int r4 = tid >> 2;            // 0..15
    int k4 = (tid & 3) * 8;       // 0,8,16,24 (bf16 units -> 16B chunks)
    const unsigned short* src = mean + ((size_t)rows0 + r4) * (RREL * HID) + step * 32 + k4;
    __builtin_amdgcn_global_load_async_to_lds_b128(
        (__attribute__((address_space(1))) v4i*)(void*)src,
        (__attribute__((address_space(3))) v4i*)(void*)(dst + r4 * 32 + k4),
        0, 0);
  };
  if (tid < 64) async_issue(0, abuf[0]);
  wait_async0();
#else
  ((unsigned int*)abuf[0])[tid] = stage_mean(0);
#endif
  __syncthreads();

  v8f acc0 = {}, acc1 = {};
  int cur = 0;
  int nt0 = wave * 2, nt1 = wave * 2 + 1;
  int row = lane & 15, kb = (lane >> 4) * 8;
  for (int step = 0; step < 56; ++step) {
    unsigned int nxt = 0;
    bool manual = false;
    if (step + 1 < 56) {
#if ASYNC_OK
      if (step + 1 < 48) {
        if (tid < 64) async_issue(step + 1, abuf[cur ^ 1]);
      } else {
        nxt = stage_h(step + 1);
        manual = true;
      }
#else
      nxt = (step + 1 < 48) ? stage_mean(step + 1) : stage_h(step + 1);
      manual = true;
#endif
    }
    Frag16 a;
    {
      const unsigned short* p = abuf[cur] + row * 32 + kb;
      a.h[0] = *(const v8bf*)p;
      a.h[1] = *(const v8bf*)(p + 16);
    }
    v16bf b0 = load_bfrag(wsw, step, nt0, lane, 16);
    v16bf b1 = load_bfrag(wsw, step, nt1, lane, 16);
    acc0 = __builtin_amdgcn_wmma_f32_16x16x32_bf16(false, a.v, false, b0, (short)0, acc0, false, false);
    acc1 = __builtin_amdgcn_wmma_f32_16x16x32_bf16(false, a.v, false, b1, (short)0, acc1, false, false);
    if (manual) ((unsigned int*)abuf[cur ^ 1])[tid] = nxt;
    wait_async0();
    __syncthreads();
    cur ^= 1;
  }

  int hi = lane >> 4, cl = lane & 15;
#pragma unroll
  for (int v = 0; v < 8; ++v) {
    int r = v + 8 * hi;
    int c0 = wave * 32 + cl, c1 = c0 + 16;
    size_t ro = (size_t)(rows0 + r) * HID;
    tile[r * HID + c0] = acc0[v] + cbias[c0] + ebias[ro + c0];
    tile[r * HID + c1] = acc1[v] + cbias[c1] + ebias[ro + c1];
  }
  __syncthreads();
#pragma unroll
  for (int rr = 0; rr < 2; ++rr) {
    int r = wave * 2 + rr;
    float s = 0.f, sq = 0.f;
    for (int c = lane; c < HID; c += 32) { float xv = tile[r * HID + c]; s += xv; sq += xv * xv; }
#pragma unroll
    for (int o = 16; o > 0; o >>= 1) { s += __shfl_xor(s, o, 32); sq += __shfl_xor(sq, o, 32); }
    if (lane == 0) {
      float m = s / 256.f;
      s_mu[r] = m;
      s_rs[r] = rsqrtf(fmaxf(sq / 256.f - m * m, 0.f) + LN_EPS);
    }
  }
  __syncthreads();
  for (int r = 0; r < 16; ++r) {
    size_t o = (size_t)(rows0 + r) * HID + tid;
    float xv = tile[r * HID + tid];
    float y = (xv - s_mu[r]) * s_rs[r] * g[tid] + b[tid];
    hout[o] = fmaxf(y, 0.f) + hin[o];
  }
}

// ---------------- launcher ----------------

extern "C" void kernel_launch(void* const* d_in, const int* in_sizes, int n_in,
                              void* d_out, int out_size, void* d_ws, size_t ws_size,
                              hipStream_t stream) {
  (void)n_in; (void)ws_size;
  const float* x0    = (const float*)d_in[0];
  const float* x1    = (const float*)d_in[1];
  const float* x2    = (const float*)d_in[2];
  const float* x3    = (const float*)d_in[3];
  const int*   esrc  = (const int*)d_in[4];
  const int*   edst  = (const int*)d_in[5];
  const int*   etyp  = (const int*)d_in[6];
  const float* ew    = (const float*)d_in[7];
  const float* projW = (const float*)d_in[8];
  const float* pbias = (const float*)d_in[9];
  const float* pg    = (const float*)d_in[10];
  const float* pb    = (const float*)d_in[11];
  const float* ete   = (const float*)d_in[12];
  const float* emlpW = (const float*)d_in[13];
  const float* emlpb = (const float*)d_in[14];
  const float* basis = (const float*)d_in[15];
  const float* comp  = (const float*)d_in[16];
  const float* root  = (const float*)d_in[17];
  const float* convb = (const float*)d_in[18];
  const float* lng   = (const float*)d_in[19];
  const float* lnb   = (const float*)d_in[20];

  const int NE = in_sizes[0] / FIN, NF = in_sizes[1] / FIN;
  const int NP = in_sizes[2] / FIN, NI = in_sizes[3] / FIN;
  const int N  = NE + NF + NP + NI;
  const int E  = in_sizes[4];
  const int L  = in_sizes[18] / HID;

  char* p = (char*)d_ws;
  auto take = [&](size_t bytes) -> void* {
    void* r = (void*)p;
    p += (bytes + 255) & ~(size_t)255;
    return r;
  };
  float* h0            = (float*)take((size_t)N * HID * 4);
  float* h1            = (float*)take((size_t)N * HID * 4);
  float* ebias         = (float*)take((size_t)N * HID * 4);
  float* deg           = (float*)take((size_t)N * 4);
  float* cnt           = (float*)take((size_t)N * RREL * 4);
  float* sums          = (float*)take((size_t)N * RREL * HID * 4);
  unsigned short* mean = (unsigned short*)take((size_t)N * RREL * HID * 2);
  unsigned short* wsw  = (unsigned short*)take((size_t)KCAT * HID * 2);
  unsigned short* pwsw = (unsigned short*)take((size_t)4 * FIN * HID * 2);
  float* relw          = (float*)take((size_t)RREL * HID * 4);

  (void)hipMemsetAsync(ebias, 0, (size_t)N * HID * 4, stream);
  (void)hipMemsetAsync(deg, 0, (size_t)N * 4, stream);
  (void)hipMemsetAsync(cnt, 0, (size_t)N * RREL * 4, stream);

  make_projw_kernel<<<(4 * FIN * HID) / 256, 256, 0, stream>>>(projW, pwsw);
  relw_kernel<<<RREL, 256, 0, stream>>>(ete, emlpW, relw);
  proj_kernel<<<N / 16, 256, 0, stream>>>(x0, x1, x2, x3, NE, NF, NP, pwsw, pbias, pg, pb, h0);
  edge_bias_kernel<<<E, 256, 0, stream>>>(edst, etyp, ew, relw, emlpW, emlpb, ebias, deg, cnt);
  finalize_bias_kernel<<<(N * HID + 255) / 256, 256, 0, stream>>>(ebias, deg, N * HID);
  inv_cnt_kernel<<<(N * RREL + 255) / 256, 256, 0, stream>>>(cnt, N * RREL);

  float* hc = h0;
  float* hn = h1;
  for (int l = 0; l < L; ++l) {
    make_wcat_kernel<<<KCAT, 256, 0, stream>>>(comp + (size_t)l * RREL * NBAS,
                                               basis + (size_t)l * NBAS * HID * HID,
                                               root + (size_t)l * HID * HID, wsw);
    (void)hipMemsetAsync(sums, 0, (size_t)N * RREL * HID * 4, stream);
    scatter_kernel<<<E, 256, 0, stream>>>(esrc, edst, etyp, hc, sums);
    size_t nm2 = (size_t)N * RREL * HID / 2;
    scale_mean_kernel<<<(unsigned)((nm2 + 255) / 256), 256, 0, stream>>>(
        sums, cnt, (unsigned int*)mean, nm2);
    rgcn_layer_kernel<<<N / 16, 256, 0, stream>>>(mean, hc, wsw, convb + (size_t)l * HID,
                                                  lng + (size_t)l * HID, lnb + (size_t)l * HID,
                                                  ebias, hn);
    float* tmp = hc; hc = hn; hn = tmp;
  }
  (void)hipMemcpyAsync(d_out, hc, (size_t)out_size * 4, hipMemcpyDeviceToDevice, stream);
}
